// Dynamic_estimator_850403524973
// MI455X (gfx1250) — matrix operand order repaired and checked by reading, not compiled, not run on hardware
//
#include <hip/hip_runtime.h>

// ---------------------------------------------------------------------------
// out[b,c] = exp(-( (x^2)@w^T - 2 x@(m*w)^T + sum_d m^2 w ))
// B=8192, C=2000, D=1024  ->  single fused GEMM, K = 2D = 2048.
// Compute-bound (~580 FLOP/byte); x + prepped weights are L2-resident (192MB).
// fp32 accuracy via bf16 hi/lo split: 3x v_wmma_f32_16x16x32_bf16 per K=32,
// ~2.7x fewer matrix ops than the f32 16x16x4 path.
// This revision adds register-stage software pipelining: global loads for
// slab k+1 are issued right after the LDS-fill barrier and overlap the
// 24-WMMA bundle of slab k (low occupancy at ~170 VGPRs makes this matter).
// ---------------------------------------------------------------------------

typedef __attribute__((ext_vector_type(16))) __bf16 v16bf;
typedef __attribute__((ext_vector_type(8)))  __bf16 v8bf;
typedef __attribute__((ext_vector_type(8)))  float  v8f;

union V16 { v16bf v; v8bf h[2]; };

#define BATCH 8192
#define NCLS  2000
#define NPAD  2048
#define DIM   1024
#define KFUSE 2048

#define BM 128
#define BN 128
#define BK 32
#define LDK 40   // padded ushort row stride in LDS (80B: spreads banks, keeps 16B align)

__device__ __forceinline__ void split_bf16(float f, unsigned short& hi, unsigned short& lo) {
    unsigned u = __float_as_uint(f);
    hi = (unsigned short)(u >> 16);                       // truncated high bf16
    float fhi = __uint_as_float(u & 0xffff0000u);
    float fl  = f - fhi;                                  // exact residual
    lo = (unsigned short)(__float_as_uint(fl) >> 16);     // residual as bf16
}

// ---------------------------------------------------------------------------
// Prep: per class c compute w = 1/(2*softplus(rho)^2), emit fused B rows
// [w | -2*m*w] as bf16 hi/lo planes (zero-padded to NPAD rows), and
// bias[c] = sum_d m^2 * w.
// ---------------------------------------------------------------------------
__global__ __launch_bounds__(256) void prep_kernel(
    const float* __restrict__ mean, const float* __restrict__ rho,
    unsigned short* __restrict__ Bhi, unsigned short* __restrict__ Blo,
    float* __restrict__ bias)
{
    const int c = blockIdx.x;
    const int t = threadIdx.x;
    __shared__ float red[256];

    if (c < NCLS) {
        const int d0 = t * 4;                                   // 256*4 = 1024 = DIM
        const float4 m4 = *(const float4*)(mean + (size_t)c * DIM + d0);
        const float4 r4 = *(const float4*)(rho  + (size_t)c * DIM + d0);
        const float mm[4] = {m4.x, m4.y, m4.z, m4.w};
        const float rr[4] = {r4.x, r4.y, r4.z, r4.w};

        unsigned short wh[4], wl[4], gh[4], gl[4];
        float part = 0.0f;
        #pragma unroll
        for (int i = 0; i < 4; ++i) {
            float sp  = log1pf(expf(rr[i]));        // softplus
            float w   = 1.0f / (2.0f * sp * sp);
            float g   = -2.0f * mm[i] * w;
            part     += mm[i] * mm[i] * w;
            split_bf16(w, wh[i], wl[i]);
            split_bf16(g, gh[i], gl[i]);
        }
        const size_t base = (size_t)c * KFUSE + d0;
        *(ushort4*)(Bhi + base)       = make_ushort4(wh[0], wh[1], wh[2], wh[3]);
        *(ushort4*)(Blo + base)       = make_ushort4(wl[0], wl[1], wl[2], wl[3]);
        *(ushort4*)(Bhi + base + DIM) = make_ushort4(gh[0], gh[1], gh[2], gh[3]);
        *(ushort4*)(Blo + base + DIM) = make_ushort4(gl[0], gl[1], gl[2], gl[3]);

        red[t] = part;
        __syncthreads();
        #pragma unroll
        for (int s = 128; s > 0; s >>= 1) {
            if (t < s) red[t] += red[t + s];
            __syncthreads();
        }
        if (t == 0) bias[c] = red[0];
    } else {
        const size_t base = (size_t)c * KFUSE + (size_t)t * 8;  // 256*8 = 2048
        *(uint4*)(Bhi + base) = make_uint4(0u, 0u, 0u, 0u);
        *(uint4*)(Blo + base) = make_uint4(0u, 0u, 0u, 0u);
        if (t == 0) bias[c] = 0.0f;
    }
}

// ---------------------------------------------------------------------------
// GEMM: 128x128 block tile, 8 waves (2 M x 4 N), each wave 64x32 output.
// Register-pipelined: global loads for slab k+1 overlap WMMAs of slab k.
// ---------------------------------------------------------------------------
__global__ __launch_bounds__(256) void gauss_gemm_kernel(
    const float* __restrict__ x,
    const unsigned short* __restrict__ Bhi,
    const unsigned short* __restrict__ Blo,
    const float* __restrict__ bias,
    float* __restrict__ out)
{
    __shared__ unsigned short Ah[BM][LDK];
    __shared__ unsigned short Al[BM][LDK];
    __shared__ unsigned short Bh[BN][LDK];
    __shared__ unsigned short Bl[BN][LDK];

    const int tid  = threadIdx.x;
    const int lane = tid & 31;
    const int wave = tid >> 5;       // 0..7
    const int wm   = wave & 1;       // M half (64 rows each)
    const int wn   = wave >> 1;      // 0..3 (32 cols each)
    const int mBase = blockIdx.x * BM;
    const int nBase = blockIdx.y * BN;

    v8f acc[4][2] = {};

    // cooperative-load mappings
    const int aQuad = tid & 7;       // float4 slot within a 32-wide K row
    const int aRow0 = tid >> 3;      // 0..31 (4 rows per thread, stride 32)
    const int bRow  = tid >> 1;      // 0..127
    const int bChunk = (tid & 1) * 16;

    // ---- register staging buffers (pipeline stage) ----
    float4 aReg[4];                  // raw x values (x^2 applied at load)
    uint4  bhReg, blReg;             // prepped bf16 hi/lo, 8 ushorts each

    // helper lambdas keep load/store code in one place
    auto loadStage = [&](int kf) {
        const bool sq   = (kf < DIM);                   // uniform per K-slab
        const int  kcol = (sq ? kf : kf - DIM) + aQuad * 4;
        #pragma unroll
        for (int i = 0; i < 4; ++i) {
            const int r = aRow0 + 32 * i;
            float4 v = *(const float4*)(x + (size_t)(mBase + r) * DIM + kcol);
            if (sq) { v.x *= v.x; v.y *= v.y; v.z *= v.z; v.w *= v.w; }
            aReg[i] = v;
        }
        const size_t goff = (size_t)(nBase + bRow) * KFUSE + kf + bChunk;
        bhReg = *(const uint4*)(Bhi + goff);
        blReg = *(const uint4*)(Blo + goff);
    };
    auto storeStage = [&]() {
        #pragma unroll
        for (int i = 0; i < 4; ++i) {
            const int r = aRow0 + 32 * i;
            unsigned short h0, l0, h1, l1, h2, l2, h3, l3;
            split_bf16(aReg[i].x, h0, l0); split_bf16(aReg[i].y, h1, l1);
            split_bf16(aReg[i].z, h2, l2); split_bf16(aReg[i].w, h3, l3);
            *(ushort4*)&Ah[r][aQuad * 4] = make_ushort4(h0, h1, h2, h3);
            *(ushort4*)&Al[r][aQuad * 4] = make_ushort4(l0, l1, l2, l3);
        }
        *(uint4*)&Bh[bRow][bChunk] = bhReg;
        *(uint4*)&Bl[bRow][bChunk] = blReg;
    };

    loadStage(0);                                        // prologue

    for (int kf = 0; kf < KFUSE; kf += BK) {
        storeStage();                                    // regs (slab k) -> LDS
        __syncthreads();

        if (kf + BK < KFUSE) loadStage(kf + BK);         // overlap with WMMAs below

        // ---- fragment loads (ISA bf16 A/B layouts, aligned 16B LDS reads) ----
        const int fl = lane & 15;
        const int fh = lane >> 4;
        V16 afh[4], afl[4], bfh[2], bfl[2];
        #pragma unroll
        for (int i = 0; i < 4; ++i) {
            const int r  = wm * 64 + i * 16 + fl;
            const int k0 = 8 * fh;                 // A: lanes<16 K=0..7,16..23; lanes>=16 +8
            afh[i].h[0] = *(const v8bf*)&Ah[r][k0];
            afh[i].h[1] = *(const v8bf*)&Ah[r][k0 + 16];
            afl[i].h[0] = *(const v8bf*)&Al[r][k0];
            afl[i].h[1] = *(const v8bf*)&Al[r][k0 + 16];
        }
        #pragma unroll
        for (int j = 0; j < 2; ++j) {
            const int n  = wn * 32 + j * 16 + fl;
            const int k0 = 8 * fh;                 // B: lanes<16 K=0..7,16..23; lanes>=16 +8
            bfh[j].h[0] = *(const v8bf*)&Bh[n][k0];
            bfh[j].h[1] = *(const v8bf*)&Bh[n][k0 + 16];
            bfl[j].h[0] = *(const v8bf*)&Bl[n][k0];
            bfl[j].h[1] = *(const v8bf*)&Bl[n][k0 + 16];
        }

        // ---- 24 WMMAs: hi*hi + hi*lo + lo*hi (~fp32 accuracy) ----
        #pragma unroll
        for (int i = 0; i < 4; ++i) {
            #pragma unroll
            for (int j = 0; j < 2; ++j) {
                acc[i][j] = __builtin_amdgcn_wmma_f32_16x16x32_bf16(
                    false, afh[i].v, false, bfh[j].v, (short)0, acc[i][j], false, false);
                acc[i][j] = __builtin_amdgcn_wmma_f32_16x16x32_bf16(
                    false, afh[i].v, false, bfl[j].v, (short)0, acc[i][j], false, false);
                acc[i][j] = __builtin_amdgcn_wmma_f32_16x16x32_bf16(
                    false, afl[i].v, false, bfh[j].v, (short)0, acc[i][j], false, false);
            }
        }
        __syncthreads();
    }

    // ---- epilogue: out = exp(-(acc + bias[c])); C/D layout: VGPR e ->
    //      row = tile + e + 8*(lane>>4), col = tile + (lane&15) ----
    const int colOff = lane & 15;
    const int rowOff = 8 * (lane >> 4);
    #pragma unroll
    for (int j = 0; j < 2; ++j) {
        const int gc = nBase + wn * 32 + j * 16 + colOff;
        if (gc < NCLS) {
            const float bb = bias[gc];
            #pragma unroll
            for (int i = 0; i < 4; ++i) {
                const int grBase = mBase + wm * 64 + i * 16 + rowOff;
                #pragma unroll
                for (int e = 0; e < 8; ++e) {
                    out[(size_t)(grBase + e) * NCLS + gc] = __expf(-(acc[i][j][e] + bb));
                }
            }
        }
    }
}

// ---------------------------------------------------------------------------
extern "C" void kernel_launch(void* const* d_in, const int* in_sizes, int n_in,
                              void* d_out, int out_size, void* d_ws, size_t ws_size,
                              hipStream_t stream) {
    const float* x    = (const float*)d_in[0];   // [8192,1024]
    const float* mean = (const float*)d_in[1];   // [2000,1024]
    const float* rho  = (const float*)d_in[2];   // [2000,1024]
    float* out = (float*)d_out;                  // [8192,2000]

    // workspace: Bhi[NPAD*KFUSE] u16 | Blo[NPAD*KFUSE] u16 | bias[NPAD] f32 (~17MB)
    unsigned short* Bhi  = (unsigned short*)d_ws;
    unsigned short* Blo  = Bhi + (size_t)NPAD * KFUSE;
    float*          bias = (float*)(Blo + (size_t)NPAD * KFUSE);

    prep_kernel<<<NPAD, 256, 0, stream>>>(mean, rho, Bhi, Blo, bias);
    gauss_gemm_kernel<<<dim3(BATCH / BM, NPAD / BN), 256, 0, stream>>>(x, Bhi, Blo, bias, out);
}
